// SparseAttention_86242943304137
// MI455X (gfx1250) — matrix operand order repaired and checked by reading, not compiled, not run on hardware
//
#include <hip/hip_runtime.h>
#include <hip/hip_bf16.h>

// Sliding-window (banded) multi-head attention for gfx1250 (MI455X).
// Pipeline: f32->f16 convert -> QKV GEMM (WMMA, scatter q/k in [B,H,N,D],
// v transposed in [B,H,D,N]) -> flash-style banded attention (WMMA) ->
// output projection GEMM (WMMA).

typedef __attribute__((ext_vector_type(16))) _Float16 v16h;
typedef __attribute__((ext_vector_type(8)))  _Float16 v8h;
typedef __attribute__((ext_vector_type(8)))  float    v8f;
typedef __attribute__((ext_vector_type(4)))  float    v4f;

#define SEQ_N   2048
#define EMB_C   1024
#define HEADS_H 16
#define HEAD_D  64
#define BATCH_B 2
#define WIN     128
#define MROWS   4096           /* B*N */
#define QKV_C   3072           /* 3*C */
#define PER_T   4194304        /* B*H*N*D = one q/k/v tensor, in halves */

// ---------- fragment loaders (contiguous k order in A and B; not the ISA table order) ----------

// A-matrix 16x32 f16: lane holds row (lane&15); lanes<16: K={0..15},
// lanes>=16: K={16..31}. Two contiguous 16B loads per lane.
__device__ inline v16h load_a_frag(const _Float16* __restrict__ tile, int ldk) {
    const int lane = threadIdx.x & 31;
    const int row  = lane & 15;
    const int kb   = (lane & 16) ? 16 : 0;
    const _Float16* p = tile + (size_t)row * ldk + kb;
    v8h lo = *(const v8h*)(p);
    v8h hi = *(const v8h*)(p + 8);
    v16h r;
#pragma unroll
    for (int i = 0; i < 8; ++i) { r[i] = lo[i]; r[i + 8] = hi[i]; }
    return r;
}

// B-matrix 32x16 f16 where B = W^T, W row-major [ncols, ldk]:
// lane -> output column n (lane&15); halves -> K = kbase + h,
// kbase = 0 (lanes<16) / 16 (lanes>=16). One contiguous 32B load per lane.
__device__ inline v16h load_bT_frag(const _Float16* __restrict__ W, int ldk,
                                    int cn, int kk) {
    const int lane = threadIdx.x & 31;
    const int n    = lane & 15;
    const int kb   = (lane & 16) ? 16 : 0;
    return *(const v16h*)(W + (size_t)(cn + n) * ldk + kk + kb);
}

// ---------- f32 -> f16 conversion (8 elems / thread) ----------
__global__ void cvt_f32_f16_kernel(const float* __restrict__ in,
                                   _Float16* __restrict__ out, int n8) {
    const int i = blockIdx.x * blockDim.x + threadIdx.x;
    if (i >= n8) return;
    const size_t o = (size_t)i * 8;
    v4f f0 = *(const v4f*)(in + o);
    v4f f1 = *(const v4f*)(in + o + 4);
    v8h r;
#pragma unroll
    for (int j = 0; j < 4; ++j) { r[j] = (_Float16)f0[j]; r[j + 4] = (_Float16)f1[j]; }
    *(v8h*)(out + o) = r;
}

// ---------- QKV GEMM: [4096,3072] = xh[4096,1024] @ wqkv[3072,1024]^T + b ----------
// Each wave: 32 rows x 64 cols (2 A-frags x 4 B-frags -> 8 WMMAs / K-step).
// Scatter: q,k -> [B,H,N,D]; v -> [B,H,D,N].
__global__ void __launch_bounds__(128)
qkv_gemm_kernel(const _Float16* __restrict__ xh,
                const _Float16* __restrict__ wh,
                const float* __restrict__ bias,
                _Float16* __restrict__ qkv) {
    const int wave = __builtin_amdgcn_readfirstlane(threadIdx.x >> 5);
    const int lane = threadIdx.x & 31;
    const int m0   = blockIdx.x * 32;
    const int c0   = blockIdx.y * 256 + wave * 64;

    v8f acc[2][4] = {};
    const _Float16* at = xh + (size_t)m0 * EMB_C;
    for (int kk = 0; kk < EMB_C; kk += 32) {
        v16h a0 = load_a_frag(at + kk, EMB_C);
        v16h a1 = load_a_frag(at + (size_t)16 * EMB_C + kk, EMB_C);
#pragma unroll
        for (int t = 0; t < 4; ++t) {
            v16h b = load_bT_frag(wh, EMB_C, c0 + t * 16, kk);
            acc[0][t] = __builtin_amdgcn_wmma_f32_16x16x32_f16(
                false, a0, false, b, (short)0, acc[0][t], false, false);
            acc[1][t] = __builtin_amdgcn_wmma_f32_16x16x32_f16(
                false, a1, false, b, (short)0, acc[1][t], false, false);
        }
    }

    const int ln   = lane & 15;
    const int hi16 = lane >> 4;
#pragma unroll
    for (int t = 0; t < 4; ++t) {
        const int c   = c0 + t * 16 + ln;      // column in [0,3072)
        const int s   = c >> 10;               // 0=q 1=k 2=v
        const int rem = c & 1023;
        const int h   = rem >> 6;
        const int d   = rem & 63;
        const float bv = bias[c];
#pragma unroll
        for (int u = 0; u < 2; ++u) {
#pragma unroll
            for (int r = 0; r < 8; ++r) {
                const int row = m0 + u * 16 + r + 8 * hi16; // in [0,4096)
                const int b   = row >> 11;
                const int nr  = row & (SEQ_N - 1);
                const int bh  = b * HEADS_H + h;
                size_t di;
                if (s == 2) {
                    // v transposed: [B,H,D,N]
                    di = 2 * (size_t)PER_T +
                         (((size_t)bh * HEAD_D + d) * SEQ_N + nr);
                } else {
                    di = (size_t)s * PER_T +
                         (((size_t)bh * SEQ_N + nr) * HEAD_D + d);
                }
                qkv[di] = (_Float16)(acc[u][t][r] + bv);
            }
        }
    }
}

// ---------- banded flash attention ----------
// One wave per 16-query tile of one (b,h). Key slabs of 32, clipped to band.
#define ATT_WAVES 4
#define LDP 40   /* padded LDS row stride (halves) */

__global__ void __launch_bounds__(ATT_WAVES * 32)
banded_attn_kernel(const _Float16* __restrict__ qkv,
                   _Float16* __restrict__ aout) {
    __shared__ _Float16 plds[ATT_WAVES][16][LDP];

    const int wave = __builtin_amdgcn_readfirstlane(threadIdx.x >> 5);
    const int lane = threadIdx.x & 31;
    const int bh   = blockIdx.x;                     // [0, 32)
    const int b    = bh >> 4;
    const int h    = bh & 15;
    const int qt   = blockIdx.y * ATT_WAVES + wave;  // [0, 128)
    const int qi   = qt * 16;                        // wave-uniform (scalar)

    const _Float16* qb = qkv + 0 * (size_t)PER_T + (size_t)bh * SEQ_N * HEAD_D;
    const _Float16* kb = qkv + 1 * (size_t)PER_T + (size_t)bh * SEQ_N * HEAD_D;
    const _Float16* vt = qkv + 2 * (size_t)PER_T + (size_t)bh * HEAD_D * SEQ_N;

    // Q fragments, K-dim = D = 64 -> two 16x32 frags; 1/sqrt(D)=2^-3 folded in.
    v16h qa0 = load_a_frag(qb + (size_t)qi * HEAD_D,      HEAD_D);
    v16h qa1 = load_a_frag(qb + (size_t)qi * HEAD_D + 32, HEAD_D);
#pragma unroll
    for (int i = 0; i < 16; ++i) {
        qa0[i] = qa0[i] * (_Float16)0.125f;
        qa1[i] = qa1[i] * (_Float16)0.125f;
    }

    v8f o[4] = {};
    float m_i[8], l_i[8];
#pragma unroll
    for (int r = 0; r < 8; ++r) { m_i[r] = -3.0e38f; l_i[r] = 0.0f; }

    const int ln   = lane & 15;
    const int hi16 = lane >> 4;

    const int jlo = (qi - WIN) > 0 ? (qi - WIN) : 0;
    const int jhi = (qi + 15 + WIN) < (SEQ_N - 1) ? (qi + 15 + WIN) : (SEQ_N - 1);

    // Note: j0 in [0, 2016], so all rows j0..j0+31 are in [0, N); only the
    // band mask zeroes out-of-window columns.
    for (int j0 = jlo & ~31; j0 <= jhi; j0 += 32) {
        // S = Q @ K^T for 32 keys: two 16x16 C-frags, each 2 WMMAs over D
        v8f s0 = {}, s1 = {};
        {
            const int kbo = (lane & 16) ? 16 : 0;
            const int j   = j0 + ln;
            const int j2  = j0 + 16 + ln;
            v16h b00 = *(const v16h*)(kb + (size_t)j  * HEAD_D + 0  + kbo);
            v16h b01 = *(const v16h*)(kb + (size_t)j  * HEAD_D + 32 + kbo);
            v16h b10 = *(const v16h*)(kb + (size_t)j2 * HEAD_D + 0  + kbo);
            v16h b11 = *(const v16h*)(kb + (size_t)j2 * HEAD_D + 32 + kbo);
            s0 = __builtin_amdgcn_wmma_f32_16x16x32_f16(false, qa0, false, b00,
                                                        (short)0, s0, false, false);
            s0 = __builtin_amdgcn_wmma_f32_16x16x32_f16(false, qa1, false, b01,
                                                        (short)0, s0, false, false);
            s1 = __builtin_amdgcn_wmma_f32_16x16x32_f16(false, qa0, false, b10,
                                                        (short)0, s1, false, false);
            s1 = __builtin_amdgcn_wmma_f32_16x16x32_f16(false, qa1, false, b11,
                                                        (short)0, s1, false, false);
        }

        // Full-tile fast path: all (i,j) in band iff qi-113 <= j0 <= qi+97
        // (wave-uniform -> scalar branch).
        const bool needmask = (j0 < qi - 113) || (j0 > qi + 97);
        if (needmask) {
#pragma unroll
            for (int r = 0; r < 8; ++r) {
                const int i = qi + r + 8 * hi16;
                const int ja  = j0 + ln;
                const int jb2 = j0 + 16 + ln;
                const bool oka = (ja  >= i - WIN) && (ja  <= i + WIN);
                const bool okb = (jb2 >= i - WIN) && (jb2 <= i + WIN);
                s0[r] = oka ? s0[r] : -1.0e30f;
                s1[r] = okb ? s1[r] : -1.0e30f;
            }
        }

        // online softmax (rows live across 16 lanes)
#pragma unroll
        for (int r = 0; r < 8; ++r) {
            const float sa = s0[r];
            const float sb = s1[r];

            float rm = fmaxf(sa, sb);
            rm = fmaxf(rm, __shfl_xor(rm, 1));
            rm = fmaxf(rm, __shfl_xor(rm, 2));
            rm = fmaxf(rm, __shfl_xor(rm, 4));
            rm = fmaxf(rm, __shfl_xor(rm, 8));

            const float mnew  = fmaxf(m_i[r], rm);
            const float alpha = __expf(m_i[r] - mnew);
            const float p0 = __expf(sa - mnew);
            const float p1 = __expf(sb - mnew);
            float rs = p0 + p1;
            rs += __shfl_xor(rs, 1);
            rs += __shfl_xor(rs, 2);
            rs += __shfl_xor(rs, 4);
            rs += __shfl_xor(rs, 8);

            l_i[r] = l_i[r] * alpha + rs;
            m_i[r] = mnew;
#pragma unroll
            for (int t = 0; t < 4; ++t) o[t][r] *= alpha;

            const int m = r + 8 * hi16;
            plds[wave][m][ln]      = (_Float16)p0;
            plds[wave][m][16 + ln] = (_Float16)p1;
        }

        // intra-wave LDS RAW fence (store -> load transpose)
        asm volatile("s_wait_dscnt 0" ::: "memory");

        // P back in A layout (16x32 f16)
        v16h pa;
        {
            const int row = lane & 15;
            const int kb2 = (lane & 16) ? 16 : 0;
            const _Float16* pp = &plds[wave][row][kb2];
            v8h lo = *(const v8h*)(pp);
            v8h hi = *(const v8h*)(pp + 8);
#pragma unroll
            for (int i2 = 0; i2 < 8; ++i2) { pa[i2] = lo[i2]; pa[i2 + 8] = hi[i2]; }
        }

        // O += P @ V  (K-dim = 32 keys, N = D in 4 tiles of 16).
        // V stored transposed [D, N]: B[k][n] = vt[(t*16+n)*N + j0 + k]
        // -> one contiguous 32B load per lane.
        {
            const int kbo = (lane & 16) ? 16 : 0;
#pragma unroll
            for (int t = 0; t < 4; ++t) {
                v16h bv = *(const v16h*)(vt + (size_t)(t * 16 + ln) * SEQ_N +
                                         j0 + kbo);
                o[t] = __builtin_amdgcn_wmma_f32_16x16x32_f16(
                    false, pa, false, bv, (short)0, o[t], false, false);
            }
        }
    }

    // epilogue: normalize and store f16 to [B, N, H*D]
#pragma unroll
    for (int r = 0; r < 8; ++r) {
        const float inv = 1.0f / l_i[r];
        const int m   = r + 8 * hi16;
        const int row = qi + m;
        _Float16* op = aout + (((size_t)b * SEQ_N + row) * EMB_C + h * HEAD_D);
#pragma unroll
        for (int t = 0; t < 4; ++t)
            op[t * 16 + ln] = (_Float16)(o[t][r] * inv);
    }
}

// ---------- projection GEMM: d_out[4096,1024] = aout @ wproj^T + b (f32 out) ----------
__global__ void __launch_bounds__(128)
proj_gemm_kernel(const _Float16* __restrict__ ah,
                 const _Float16* __restrict__ wh,
                 const float* __restrict__ bias,
                 float* __restrict__ out) {
    const int wave = __builtin_amdgcn_readfirstlane(threadIdx.x >> 5);
    const int lane = threadIdx.x & 31;
    const int m0   = blockIdx.x * 32;
    const int c0   = blockIdx.y * 256 + wave * 64;

    v8f acc[2][4] = {};
    const _Float16* at = ah + (size_t)m0 * EMB_C;
    for (int kk = 0; kk < EMB_C; kk += 32) {
        v16h a0 = load_a_frag(at + kk, EMB_C);
        v16h a1 = load_a_frag(at + (size_t)16 * EMB_C + kk, EMB_C);
#pragma unroll
        for (int t = 0; t < 4; ++t) {
            v16h b = load_bT_frag(wh, EMB_C, c0 + t * 16, kk);
            acc[0][t] = __builtin_amdgcn_wmma_f32_16x16x32_f16(
                false, a0, false, b, (short)0, acc[0][t], false, false);
            acc[1][t] = __builtin_amdgcn_wmma_f32_16x16x32_f16(
                false, a1, false, b, (short)0, acc[1][t], false, false);
        }
    }

    const int ln   = lane & 15;
    const int hi16 = lane >> 4;
#pragma unroll
    for (int t = 0; t < 4; ++t) {
        const int c = c0 + t * 16 + ln;
        const float bv = bias[c];
#pragma unroll
        for (int u = 0; u < 2; ++u) {
#pragma unroll
            for (int r = 0; r < 8; ++r) {
                const int row = m0 + u * 16 + r + 8 * hi16;
                out[(size_t)row * EMB_C + c] = acc[u][t][r] + bv;
            }
        }
    }
}

// ---------- launch ----------
extern "C" void kernel_launch(void* const* d_in, const int* in_sizes, int n_in,
                              void* d_out, int out_size, void* d_ws, size_t ws_size,
                              hipStream_t stream) {
    const float* x      = (const float*)d_in[0];   // [2,2048,1024]
    const float* qkv_w  = (const float*)d_in[1];   // [3072,1024]
    const float* qkv_b  = (const float*)d_in[2];   // [3072]
    const float* proj_w = (const float*)d_in[3];   // [1024,1024]
    const float* proj_b = (const float*)d_in[4];   // [1024]
    float* out = (float*)d_out;

    // workspace layout (halves)
    _Float16* ws = (_Float16*)d_ws;
    const size_t N_X    = (size_t)MROWS * EMB_C;     // 4M
    const size_t N_WQKV = (size_t)QKV_C * EMB_C;     // 3M
    const size_t N_WPRJ = (size_t)EMB_C * EMB_C;     // 1M
    _Float16* xh    = ws;
    _Float16* wqkvh = xh + N_X;
    _Float16* wprjh = wqkvh + N_WQKV;
    _Float16* qkvh  = wprjh + N_WPRJ;                // 3 * PER_T halves
    _Float16* aouth = qkvh + 3 * (size_t)PER_T;

    // 1) f32 -> f16 conversions (8 elems / thread)
    {
        int n0 = (int)(N_X / 8), n1 = (int)(N_WQKV / 8), n2 = (int)(N_WPRJ / 8);
        cvt_f32_f16_kernel<<<(n0 + 255) / 256, 256, 0, stream>>>(x, xh, n0);
        cvt_f32_f16_kernel<<<(n1 + 255) / 256, 256, 0, stream>>>(qkv_w, wqkvh, n1);
        cvt_f32_f16_kernel<<<(n2 + 255) / 256, 256, 0, stream>>>(proj_w, wprjh, n2);
    }

    // 2) QKV GEMM + scatter (q,k row-major; v transposed)
    {
        dim3 grid(MROWS / 32, QKV_C / 256);
        qkv_gemm_kernel<<<grid, 128, 0, stream>>>(xh, wqkvh, qkv_b, qkvh);
    }

    // 3) banded flash attention
    {
        dim3 grid(BATCH_B * HEADS_H, SEQ_N / (16 * ATT_WAVES));
        banded_attn_kernel<<<grid, ATT_WAVES * 32, 0, stream>>>(qkvh, aouth);
    }

    // 4) output projection
    {
        dim3 grid(MROWS / 32, EMB_C / 256);
        proj_gemm_kernel<<<grid, 128, 0, stream>>>(aouth, wprjh, proj_b, out);
    }
}
